// TemporalContrastiveLoss_27779848470677
// MI455X (gfx1250) — matrix operand order repaired and checked by reading, not compiled, not run on hardware
//
#include <hip/hip_runtime.h>

typedef __attribute__((ext_vector_type(16))) _Float16 v16h;
typedef __attribute__((ext_vector_type(8)))  _Float16 v8h;
typedef __attribute__((ext_vector_type(8)))  float    v8f;
typedef __attribute__((ext_vector_type(4)))  float    v4f;

#define TT     256   // T
#define DD     256   // D
#define NB     512   // B
#define PITCH  264   // halves per LDS row (256 + 8 pad -> 528B pitch, kills bank conflicts)

__device__ __forceinline__ v16h cat16(v8h lo, v8h hi) {
  union { v16h w; v8h h[2]; } u;
  u.h[0] = lo; u.h[1] = hi;
  return u.w;
}

__global__ void __launch_bounds__(64) tcl_zero_kernel(float* out) {
  if (threadIdx.x == 0) out[0] = 0.0f;
}

__global__ void __launch_bounds__(256, 2)
TemporalContrastiveLoss_27779848470677_kernel(const float* __restrict__ feat,
                                              float* __restrict__ out) {
  extern __shared__ _Float16 lds[];  // [TT][PITCH] normalized f16 rows of this batch

  const int tid  = threadIdx.x;
  const int lane = tid & 31;
  const int wave = tid >> 5;
  const float* fb = feat + (size_t)blockIdx.x * (TT * DD);

  // ---------------- Phase A: normalize rows -> f16 in LDS ----------------
  for (int row = wave; row < TT; row += 8) {
    const float* rp = fb + row * DD;
    v4f x0 = *(const v4f*)(rp + lane * 8);
    v4f x1 = *(const v4f*)(rp + lane * 8 + 4);
    float ss = x0.x*x0.x + x0.y*x0.y + x0.z*x0.z + x0.w*x0.w
             + x1.x*x1.x + x1.y*x1.y + x1.z*x1.z + x1.w*x1.w;
    #pragma unroll
    for (int m = 1; m < 32; m <<= 1) ss += __shfl_xor(ss, m, 32);
    const float invn = 1.0f / fmaxf(sqrtf(ss), 1e-12f);
    v8h hv;
    hv[0] = (_Float16)(x0.x * invn); hv[1] = (_Float16)(x0.y * invn);
    hv[2] = (_Float16)(x0.z * invn); hv[3] = (_Float16)(x0.w * invn);
    hv[4] = (_Float16)(x1.x * invn); hv[5] = (_Float16)(x1.y * invn);
    hv[6] = (_Float16)(x1.z * invn); hv[7] = (_Float16)(x1.w * invn);
    *(v8h*)(lds + row * PITCH + lane * 8) = hv;   // 16B ds_store
  }
  __syncthreads();

  // ---------------- Phase B: WMMA GEMM + fixed-shift logsumexp ----------------
  // logits = 10*dot, dot in [-1,1]  =>  exp(logit - 10) in [2e-9, 1]: no max tracking needed.
  const int N     = lane & 15;          // column-in-tile this lane owns (B/C/D layout)
  const int hiH   = (lane >> 4) & 1;    // lane group
  const int Mbase = hiH ? 8 : 0;        // C/D: vgpr v holds row M = Mbase + v

  float loss_acc = 0.0f;

  // B fragments depend only on (nt, kk): base pointer is nq-independent, so the
  // rotating prefetch below stays valid across both q-tile iterations.
  const _Float16* bbase = lds + N * PITCH + (hiH ? 16 : 0);
  v8h blo = *(const v8h*)(bbase);       // prefetch nt=0, kk=0
  v8h bhi = *(const v8h*)(bbase + 8);

  for (int nq = wave; nq < 16; nq += 8) {
    // Hoist all 8 A-fragments for this q-tile (A layout: lanes 0-15 K in {0..15},
    // lanes 16-31 K in {16..31}; row M = lane&15)
    const _Float16* ap = lds + (16 * nq + (lane & 15)) * PITCH + (hiH ? 16 : 0);
    v16h afrag[8];
    #pragma unroll
    for (int kk = 0; kk < 8; ++kk) {
      v8h lo = *(const v8h*)(ap + 32 * kk);        // K = off .. off+7
      v8h hi = *(const v8h*)(ap + 32 * kk + 8);    // K = off+8 .. off+15
      afrag[kk] = cat16(lo, hi);
    }

    float ssum[8], tgt[8];
    #pragma unroll
    for (int v = 0; v < 8; ++v) { ssum[v] = 0.0f; tgt[v] = 0.0f; }

    for (int nt = 0; nt < 16; ++nt) {
      const _Float16* bp = bbase + nt * (16 * PITCH);
      v8f acc = {};
      #pragma unroll
      for (int kk = 0; kk < 8; ++kk) {
        v16h bfrag = cat16(blo, bhi);
        // prefetch next pair: next kk, else first pair of next nt (wraps to nt=0)
        const _Float16* np = (kk < 7) ? (bp + 32 * (kk + 1))
                                      : (nt < 15 ? bp + 16 * PITCH : bbase);
        blo = *(const v8h*)(np);
        bhi = *(const v8h*)(np + 8);
        acc = __builtin_amdgcn_wmma_f32_16x16x32_f16(
                  false, afrag[kk], false, bfrag, (short)0, acc, false, false);
      }
      // shuffle-free epilogue: per-lane partial sums with fixed shift of 10
      const int t = 16 * nt + N;
      #pragma unroll
      for (int v = 0; v < 8; ++v) {
        const float ls = fmaf(acc[v], 10.0f, -10.0f);   // logit - 10
        ssum[v] += __expf(ls);
        const int q = 16 * nq + Mbase + v;
        tgt[v] += (t == q + 1) ? ls : 0.0f;             // shifted target (shift cancels below)
      }
    }

    // one cross-lane reduction per q-tile (16-lane N-group)
    #pragma unroll
    for (int v = 0; v < 8; ++v) {
      ssum[v] += __shfl_xor(ssum[v], 1, 32);
      ssum[v] += __shfl_xor(ssum[v], 2, 32);
      ssum[v] += __shfl_xor(ssum[v], 4, 32);
      ssum[v] += __shfl_xor(ssum[v], 8, 32);
      tgt[v]  += __shfl_xor(tgt[v], 1, 32);
      tgt[v]  += __shfl_xor(tgt[v], 2, 32);
      tgt[v]  += __shfl_xor(tgt[v], 4, 32);
      tgt[v]  += __shfl_xor(tgt[v], 8, 32);
    }
    if (N == 0) {   // one lane per group emits per-row loss: (log(s)+10) - (tgt+10)
      #pragma unroll
      for (int v = 0; v < 8; ++v) {
        const int q = 16 * nq + Mbase + v;
        if (q < TT - 1) loss_acc += __logf(ssum[v]) - tgt[v];
      }
    }
  }

  // wave reduction + single atomic per wave
  loss_acc += __shfl_xor(loss_acc, 16, 32);
  loss_acc += __shfl_xor(loss_acc, 8, 32);
  loss_acc += __shfl_xor(loss_acc, 4, 32);
  loss_acc += __shfl_xor(loss_acc, 2, 32);
  loss_acc += __shfl_xor(loss_acc, 1, 32);
  if (lane == 0)
    atomicAdd(out, loss_acc * (1.0f / ((float)NB * (float)(TT - 1))));
}

extern "C" void kernel_launch(void* const* d_in, const int* in_sizes, int n_in,
                              void* d_out, int out_size, void* d_ws, size_t ws_size,
                              hipStream_t stream) {
  (void)in_sizes; (void)n_in; (void)d_ws; (void)ws_size; (void)out_size;
  const float* feat = (const float*)d_in[0];
  float* out = (float*)d_out;

  const size_t lds_bytes = (size_t)TT * PITCH * sizeof(_Float16);  // 132 KB < 320 KB/WGP
  (void)hipFuncSetAttribute((const void*)TemporalContrastiveLoss_27779848470677_kernel,
                            hipFuncAttributeMaxDynamicSharedMemorySize, (int)lds_bytes);

  tcl_zero_kernel<<<1, 64, 0, stream>>>(out);
  TemporalContrastiveLoss_27779848470677_kernel<<<NB, 256, lds_bytes, stream>>>(feat, out);
}
